// EchoEA_51032801411196
// MI455X (gfx1250) — matrix operand order repaired and checked by reading, not compiled, not run on hardware
//
#include <hip/hip_runtime.h>
#include <cstddef>
#include <cstdint>

// ---------------------------------------------------------------------------
// Types for CDNA5 WMMA (wave32, 16x16x32 bf16 -> f32)
// ---------------------------------------------------------------------------
typedef __bf16 bf16;
typedef bf16  v16bf __attribute__((ext_vector_type(16)));
typedef bf16  v8bf  __attribute__((ext_vector_type(8)));
typedef float v8f   __attribute__((ext_vector_type(8)));

#define N_NODES 20000
#define N_EDGE  150000
#define N_EDGEA 300000
#define N_REL   1000
#define EH      300
#define KPAD    320   // K=300 padded to multiple of 32
#define MPAD    304   // M=300 padded to multiple of 16
#define DCAT    900
#define LD_OUT  1800

// ---------------------------------------------------------------------------
// Device helpers
// ---------------------------------------------------------------------------
__device__ __forceinline__ unsigned mapf(float f) {
  unsigned u = __float_as_uint(f);
  return (u & 0x80000000u) ? ~u : (u | 0x80000000u);
}
__device__ __forceinline__ float unmapf(unsigned u) {
  unsigned b = (u & 0x80000000u) ? (u & 0x7fffffffu) : ~u;
  return __uint_as_float(b);
}
__device__ __forceinline__ float lrelu(float x) { return x > 0.f ? x : 0.01f * x; }
__device__ __forceinline__ float sigmoidf(float x) { return 1.f / (1.f + expf(-x)); }
__device__ __forceinline__ float wave_sum(float v) {
  #pragma unroll
  for (int off = 16; off > 0; off >>= 1) v += __shfl_down(v, off, 32);
  return v;
}

// ---------------------------------------------------------------------------
// Elementwise / fill kernels
// ---------------------------------------------------------------------------
__global__ void k_fill_f32(float* __restrict__ p, float v, long n) {
  for (long i = blockIdx.x * (long)blockDim.x + threadIdx.x; i < n;
       i += (long)gridDim.x * blockDim.x) p[i] = v;
}
__global__ void k_fill_u32(unsigned* __restrict__ p, unsigned v, long n) {
  for (long i = blockIdx.x * (long)blockDim.x + threadIdx.x; i < n;
       i += (long)gridDim.x * blockDim.x) p[i] = v;
}
__global__ void k_relu(float* __restrict__ p, long n) {
  for (long i = blockIdx.x * (long)blockDim.x + threadIdx.x; i < n;
       i += (long)gridDim.x * blockDim.x) p[i] = fmaxf(p[i], 0.f);
}
__global__ void k_relu_strided(float* __restrict__ p, int rows, int F, int ld, int c0) {
  long total = (long)rows * F;
  for (long i = blockIdx.x * (long)blockDim.x + threadIdx.x; i < total;
       i += (long)gridDim.x * blockDim.x) {
    int r = (int)(i / F), f = (int)(i % F);
    size_t idx = (size_t)r * ld + c0 + f;
    p[idx] = fmaxf(p[idx], 0.f);
  }
}
__global__ void k_copy_cols(const float* __restrict__ src, float* __restrict__ dst,
                            int rows, int F, int ldDst, int c0) {
  long total = (long)rows * F;
  for (long i = blockIdx.x * (long)blockDim.x + threadIdx.x; i < total;
       i += (long)gridDim.x * blockDim.x) {
    int r = (int)(i / F), f = (int)(i % F);
    dst[(size_t)r * ldDst + c0 + f] = src[i];
  }
}

// out = g*xb + (1-g)*xa,  g = sigmoid(G + bias)
__global__ void k_highway(const float* __restrict__ G, const float* __restrict__ bias,
                          const float* __restrict__ xa, const float* __restrict__ xb,
                          float* __restrict__ out, int rows, int F, int ldOut, int c0) {
  long total = (long)rows * F;
  for (long i = blockIdx.x * (long)blockDim.x + threadIdx.x; i < total;
       i += (long)gridDim.x * blockDim.x) {
    int r = (int)(i / F), f = (int)(i % F);
    float g = sigmoidf(G[i] + bias[f]);
    out[(size_t)r * ldOut + c0 + f] = g * xb[i] + (1.f - g) * xa[i];
  }
}

// ---------------------------------------------------------------------------
// Row normalize: y = x / max(||x||, 1e-12), one wave per row
// ---------------------------------------------------------------------------
__global__ void k_rownorm(const float* __restrict__ x, float* __restrict__ y) {
  int wave = (blockIdx.x * blockDim.x + threadIdx.x) >> 5;
  int lane = threadIdx.x & 31;
  if (wave >= N_NODES) return;
  const float* xr = x + (size_t)wave * EH;
  float s = 0.f;
  for (int f = lane; f < EH; f += 32) { float v = xr[f]; s += v * v; }
  s = wave_sum(s);
  s = __shfl(s, 0, 32);
  float scale = 1.f / fmaxf(sqrtf(s), 1e-12f);
  float* yr = y + (size_t)wave * EH;
  for (int f = lane; f < EH; f += 32) yr[f] = xr[f] * scale;
}

// ---------------------------------------------------------------------------
// f32 -> bf16 with zero padding (used for X [rows,KPAD] and W [MPAD,KPAD])
// ---------------------------------------------------------------------------
__global__ void k_conv_bf16(const float* __restrict__ src, bf16* __restrict__ dst,
                            int srcRows, int srcK, int dstRows, int dstK) {
  long total = (long)dstRows * dstK;
  for (long i = blockIdx.x * (long)blockDim.x + threadIdx.x; i < total;
       i += (long)gridDim.x * blockDim.x) {
    int r = (int)(i / dstK), k = (int)(i % dstK);
    float v = (r < srcRows && k < srcK) ? src[(long)r * srcK + k] : 0.f;
    dst[i] = (bf16)v;
  }
}

// ---------------------------------------------------------------------------
// WMMA GEMM: Y[r,m] = sum_k X[r,k] * W[m,k]
//   X: [N_NODES, KPAD] bf16, W: [MPAD, KPAD] bf16, Y: [N_NODES, ldY] f32
//   One wave computes one 16x16 tile via v_wmma_f32_16x16x32_bf16.
// A-fragment (ISA 7.12.2, 16-bit A 16x32): lane L holds row (L&15);
//   lanes 0-15 K 0..15, lanes 16-31 K 16..31 (contiguous order, not the ISA table's).
// B-fragment: lane L holds column (L&15); lanes 0-15 K 0..15, lanes 16-31 K 16..31.
// ---------------------------------------------------------------------------
__global__ __launch_bounds__(256) void k_gemm_wmma(
    const bf16* __restrict__ X, const bf16* __restrict__ W, float* __restrict__ Y,
    int M, int ldY, int tilesM, int totalTiles) {
  int wave = (blockIdx.x * blockDim.x + threadIdx.x) >> 5;
  if (wave >= totalTiles) return;           // whole-wave uniform: EXEC stays all-ones
  int lane = threadIdx.x & 31;
  int tm = wave % tilesM, tn = wave / tilesM;
  int r0 = tn << 4, m0 = tm << 4;
  int half = lane >> 4;
  int arow = lane & 15, khalf = half << 3;   // 0 or 8
  int bcol = lane & 15, kb = half << 4;      // 0 or 16
  const bf16* Xr = X + (size_t)(r0 + arow) * KPAD;
  const bf16* Wr = W + (size_t)(m0 + bcol) * KPAD;
  v8f acc = {};
  #pragma unroll
  for (int kc = 0; kc < KPAD; kc += 32) {
    v8bf alo = *(const v8bf*)(Xr + kc + kb);          // K = kc+kb .. +7
    v8bf ahi = *(const v8bf*)(Xr + kc + kb + 8);      // K = kc+kb+8 .. +15
    v16bf a, b;
    #pragma unroll
    for (int i = 0; i < 8; ++i) { a[i] = alo[i]; a[i + 8] = ahi[i]; }
    b = *(const v16bf*)(Wr + kc + kb);                // 16 contiguous K per lane
    acc = __builtin_amdgcn_wmma_f32_16x16x32_bf16(
        false, a, false, b, (short)0, acc, false, false);
  }
  int ccol = lane & 15, crow = half << 3;   // C/D: VGPR v -> row crow+v, col = lane&15
  if (m0 + ccol < M) {
    #pragma unroll
    for (int v = 0; v < 8; ++v)
      Y[(size_t)(r0 + crow + v) * ldY + m0 + ccol] = acc[v];
  }
}

// ---------------------------------------------------------------------------
// Dual row-dot: sa[r] = X[r,:]·va, sb[r] = X[r,:]·vb  (one wave per row)
// ---------------------------------------------------------------------------
__global__ void k_rowdot2(const float* __restrict__ X, int ld, int F,
                          const float* __restrict__ va, const float* __restrict__ vb,
                          float* __restrict__ sa, float* __restrict__ sb, int rows) {
  int wave = (blockIdx.x * blockDim.x + threadIdx.x) >> 5;
  int lane = threadIdx.x & 31;
  if (wave >= rows) return;
  const float* xr = X + (size_t)wave * ld;
  float s0 = 0.f, s1 = 0.f;
  for (int f = lane; f < F; f += 32) { float v = xr[f]; s0 += v * va[f]; s1 += v * vb[f]; }
  s0 = wave_sum(s0);
  s1 = wave_sum(s1);
  if (lane == 0) { sa[wave] = s0; sb[wave] = s1; }
}

// ---------------------------------------------------------------------------
// GCN degree / normalization / scatter
// ---------------------------------------------------------------------------
__global__ void k_deg(const int* __restrict__ dst, float* __restrict__ deg, int n) {
  int i = blockIdx.x * blockDim.x + threadIdx.x;
  if (i < n) atomicAdd(deg + dst[i], 1.f);
}
__global__ void k_dis(float* __restrict__ d, int n) {
  int i = blockIdx.x * blockDim.x + threadIdx.x;
  if (i < n) { float v = d[i]; d[i] = v > 0.f ? rsqrtf(v) : 0.f; }
}
__global__ void k_scatter_gcn(const float* __restrict__ Xw, const int* __restrict__ src,
                              const int* __restrict__ dst, const float* __restrict__ dis,
                              float* __restrict__ out, int nE) {
  int e = blockIdx.x;
  if (e >= nE) return;
  int s = src[e], d = dst[e];
  float coef = dis[s] * dis[d];
  const float* xr = Xw + (size_t)s * EH;
  float* orow = out + (size_t)d * EH;
  for (int f = threadIdx.x; f < EH; f += blockDim.x)
    atomicAdd(orow + f, coef * xr[f]);
}

// ---------------------------------------------------------------------------
// Segment-softmax pieces
// ---------------------------------------------------------------------------
__global__ void k_edge_logit_max(const float* __restrict__ sA, const int* __restrict__ iA,
                                 const float* __restrict__ sB, const int* __restrict__ iB,
                                 const int* __restrict__ seg, float* __restrict__ l,
                                 unsigned* __restrict__ m, int n) {
  int e = blockIdx.x * blockDim.x + threadIdx.x;
  if (e >= n) return;
  float v = lrelu(sA[iA[e]] + sB[iB[e]]);
  l[e] = v;
  atomicMax(m + seg[e], mapf(v));
}
__global__ void k_edge_expsum(float* __restrict__ w, const int* __restrict__ seg,
                              const unsigned* __restrict__ m, float* __restrict__ denom,
                              int n) {
  int e = blockIdx.x * blockDim.x + threadIdx.x;
  if (e >= n) return;
  int s = seg[e];
  float ww = expf(w[e] - unmapf(m[s]));
  w[e] = ww;
  atomicAdd(denom + s, ww);
}
// out[iDst[e], c0:c0+F] += (w[e]/denom[seg[e]]) * X[iSrc[e], 0:F]
__global__ void k_scatter_soft(const float* __restrict__ X, int ldX, int F,
                               const int* __restrict__ iSrc, const int* __restrict__ iDst,
                               const int* __restrict__ seg, const float* __restrict__ w,
                               const float* __restrict__ denom, float* __restrict__ out,
                               int ldOut, int c0, int nE) {
  int e = blockIdx.x;
  if (e >= nE) return;
  int s = iSrc[e], d = iDst[e];
  float coef = w[e] / denom[seg[e]];
  const float* xr = X + (size_t)s * ldX;
  float* orow = out + (size_t)d * ldOut + c0;
  for (int f = threadIdx.x; f < F; f += blockDim.x)
    atomicAdd(orow + f, coef * xr[f]);
}

// ---------------------------------------------------------------------------
// Host orchestration
// ---------------------------------------------------------------------------
extern "C" void kernel_launch(void* const* d_in, const int* in_sizes, int n_in,
                              void* d_out, int out_size, void* d_ws, size_t ws_size,
                              hipStream_t stream) {
  (void)in_sizes; (void)n_in; (void)out_size; (void)ws_size;

  const float* x_e  = (const float*)d_in[0];
  const int*   h    = (const int*)d_in[1];
  const int*   t    = h + N_EDGE;
  const int*   rel  = (const int*)d_in[2];
  const int*   srcA = (const int*)d_in[3];
  const int*   dstA = srcA + N_EDGEA;
  // d_in[4] = rel_all (unused)
  const float* gcn1_W  = (const float*)d_in[5];
  const float* hw1_W   = (const float*)d_in[6];
  const float* hw1_b   = (const float*)d_in[7];
  const float* gat1_ai = (const float*)d_in[8];
  const float* gat1_aj = (const float*)d_in[9];
  const float* ghw1_W  = (const float*)d_in[10];
  const float* ghw1_b  = (const float*)d_in[11];
  const float* gat2_ai = (const float*)d_in[12];
  const float* gat2_aj = (const float*)d_in[13];
  const float* ghw2_W  = (const float*)d_in[14];
  const float* ghw2_b  = (const float*)d_in[15];
  const float* e2r_ah1 = (const float*)d_in[16];
  const float* e2r_ah2 = (const float*)d_in[17];
  const float* e2r_at1 = (const float*)d_in[18];
  const float* e2r_at2 = (const float*)d_in[19];
  const float* e2r_wh  = (const float*)d_in[20];
  const float* e2r_wt  = (const float*)d_in[21];
  const float* r2e_ah  = (const float*)d_in[22];
  const float* r2e_at  = (const float*)d_in[23];
  const float* r2e_ar  = (const float*)d_in[24];
  const float* r2et_ah = (const float*)d_in[25];
  const float* r2et_at = (const float*)d_in[26];
  const float* r2et_ar = (const float*)d_in[27];
  const float* hw3_W   = (const float*)d_in[28];
  const float* hw3_b   = (const float*)d_in[29];
  const float* hw4_W   = (const float*)d_in[30];
  const float* hw4_b   = (const float*)d_in[31];
  const float* gat_ai  = (const float*)d_in[32];
  const float* gat_aj  = (const float*)d_in[33];
  // d_in[34] = n_rels scalar (== N_REL)

  float* dout = (float*)d_out;

  // ---- workspace carve-out (bump allocator, 256B aligned) ----
  char* ws = (char*)d_ws;
  size_t off = 0;
  auto alloc = [&](size_t bytes) -> char* {
    char* p = ws + off;
    off += (bytes + 255) & ~(size_t)255;
    return p;
  };
  const size_t MAT = sizeof(float) * (size_t)N_NODES * EH;
  float* S1 = (float*)alloc(MAT);   // x0 -> x2 -> xrh -> xh2
  float* S2 = (float*)alloc(MAT);   // x1 -> xrt -> xt2
  float* S3 = (float*)alloc(MAT);   // xe
  float* S4 = (float*)alloc(MAT);   // tmpG (GEMM out / highway gate)
  float* S5 = (float*)alloc(MAT);   // agg -> xh1
  float* S6 = (float*)alloc(MAT);   // xt1
  bf16* xbf = (bf16*)alloc(sizeof(bf16) * (size_t)N_NODES * KPAD);
  bf16* wbf = (bf16*)alloc(sizeof(bf16) * (size_t)MPAD * KPAD);
  float* xr_h  = (float*)alloc(sizeof(float) * (size_t)N_REL * EH);
  float* xr_t  = (float*)alloc(sizeof(float) * (size_t)N_REL * EH);
  float* ew    = (float*)alloc(sizeof(float) * N_EDGEA);
  unsigned* mseg = (unsigned*)alloc(sizeof(unsigned) * N_NODES);
  float* denom = (float*)alloc(sizeof(float) * N_NODES);
  float* sv0 = (float*)alloc(sizeof(float) * N_NODES);
  float* sv1 = (float*)alloc(sizeof(float) * N_NODES);
  float* sv2 = (float*)alloc(sizeof(float) * N_NODES);
  float* sv3 = (float*)alloc(sizeof(float) * N_NODES);
  float* degdis = (float*)alloc(sizeof(float) * N_NODES);

  const int TB = 256;
  auto cdiv = [](long a, long b) { return (int)((a + b - 1) / b); };

  const int tilesM = (EH + 15) / 16;                 // 19
  const int totalTiles = (N_NODES / 16) * tilesM;    // 23750
  const int gemmBlocks = cdiv(totalTiles, TB / 32);  // 8 waves / block

  auto gemm = [&](const float* X, const float* W, float* Y) {
    k_conv_bf16<<<4096, TB, 0, stream>>>(X, xbf, N_NODES, EH, N_NODES, KPAD);
    k_conv_bf16<<<400, TB, 0, stream>>>(W, wbf, EH, EH, MPAD, KPAD);
    k_gemm_wmma<<<gemmBlocks, TB, 0, stream>>>(xbf, wbf, Y, EH, EH, tilesM, totalTiles);
  };
  auto soft_edges = [&](const float* sa, const int* ia, const float* sb, const int* ib,
                        const int* seg, int nSeg, int nE) {
    k_fill_u32<<<cdiv(nSeg, TB), TB, 0, stream>>>(mseg, 0x007FFFFFu, nSeg); // map(-inf)
    k_fill_f32<<<cdiv(nSeg, TB), TB, 0, stream>>>(denom, 0.f, nSeg);
    k_edge_logit_max<<<cdiv(nE, TB), TB, 0, stream>>>(sa, ia, sb, ib, seg, ew, mseg, nE);
    k_edge_expsum<<<cdiv(nE, TB), TB, 0, stream>>>(ew, seg, mseg, denom, nE);
  };
  const long NE300 = (long)N_NODES * EH;
  const int rowBlocks = cdiv((long)N_NODES * 32, TB);

  // ---- 1) x0 = row-normalize(x_e) ----
  float* x0 = S1;
  k_rownorm<<<rowBlocks, TB, 0, stream>>>(x_e, x0);

  // ---- 2) GCN layer on edge_index_all ----
  gemm(x0, gcn1_W, S4);                                       // S4 = x0 @ gcn1_W^T
  k_fill_f32<<<cdiv(N_NODES, TB), TB, 0, stream>>>(degdis, 0.f, N_NODES);
  k_deg<<<cdiv(N_EDGEA, TB), TB, 0, stream>>>(dstA, degdis, N_EDGEA);
  k_dis<<<cdiv(N_NODES, TB), TB, 0, stream>>>(degdis, N_NODES);
  k_fill_f32<<<8192, TB, 0, stream>>>(S5, 0.f, NE300);
  k_scatter_gcn<<<N_EDGEA, 128, 0, stream>>>(S4, srcA, dstA, degdis, S5, N_EDGEA);
  k_relu<<<8192, TB, 0, stream>>>(S5, NE300);

  // ---- 3) x1 = highway(x0, gcn, hw1) ----
  gemm(x0, hw1_W, S4);
  float* x1 = S2;
  k_highway<<<8192, TB, 0, stream>>>(S4, hw1_b, x0, S5, x1, N_NODES, EH, EH, 0);

  // ---- 4) gat1 on x1, then x2 = highway(x1, gat1, ghw1) ----
  k_rowdot2<<<rowBlocks, TB, 0, stream>>>(x1, EH, EH, gat1_ai, gat1_aj, sv0, sv1, N_NODES);
  soft_edges(sv0, dstA, sv1, srcA, dstA, N_NODES, N_EDGEA);
  k_fill_f32<<<8192, TB, 0, stream>>>(S5, 0.f, NE300);
  k_scatter_soft<<<N_EDGEA, 128, 0, stream>>>(x1, EH, EH, srcA, dstA, dstA, ew, denom,
                                              S5, EH, 0, N_EDGEA);
  k_relu<<<8192, TB, 0, stream>>>(S5, NE300);
  gemm(x1, ghw1_W, S4);
  float* x2 = S1;  // x0 dead
  k_highway<<<8192, TB, 0, stream>>>(S4, ghw1_b, x1, S5, x2, N_NODES, EH, EH, 0);

  // ---- 5) gat2 on x2, then xe = highway(x1, gat2, ghw2) ----
  k_rowdot2<<<rowBlocks, TB, 0, stream>>>(x2, EH, EH, gat2_ai, gat2_aj, sv0, sv1, N_NODES);
  soft_edges(sv0, dstA, sv1, srcA, dstA, N_NODES, N_EDGEA);
  k_fill_f32<<<8192, TB, 0, stream>>>(S5, 0.f, NE300);
  k_scatter_soft<<<N_EDGEA, 128, 0, stream>>>(x2, EH, EH, srcA, dstA, dstA, ew, denom,
                                              S5, EH, 0, N_EDGEA);
  k_relu<<<8192, TB, 0, stream>>>(S5, NE300);
  gemm(x1, ghw2_W, S4);
  float* xe = S3;
  k_highway<<<8192, TB, 0, stream>>>(S4, ghw2_b, x1, S5, xe, N_NODES, EH, EH, 0);

  // ---- 6) E->R attention: xrh/xrt projections, x_r_h / x_r_t ----
  float* xrh = S1;  // x2 dead
  float* xrt = S2;  // x1 dead
  gemm(xe, e2r_wh, xrh);
  gemm(xe, e2r_wt, xrt);
  k_rowdot2<<<rowBlocks, TB, 0, stream>>>(xrh, EH, EH, e2r_ah1, e2r_at1, sv0, sv1, N_NODES);
  k_rowdot2<<<rowBlocks, TB, 0, stream>>>(xrt, EH, EH, e2r_ah2, e2r_at2, sv2, sv3, N_NODES);
  const long RE300 = (long)N_REL * EH;
  soft_edges(sv0, h, sv2, t, rel, N_REL, N_EDGE);             // e1 over rel segments
  k_fill_f32<<<cdiv(RE300, TB), TB, 0, stream>>>(xr_h, 0.f, RE300);
  k_scatter_soft<<<N_EDGE, 128, 0, stream>>>(xrh, EH, EH, h, rel, rel, ew, denom,
                                             xr_h, EH, 0, N_EDGE);
  soft_edges(sv1, h, sv3, t, rel, N_REL, N_EDGE);             // e2 over rel segments
  k_fill_f32<<<cdiv(RE300, TB), TB, 0, stream>>>(xr_t, 0.f, RE300);
  k_scatter_soft<<<N_EDGE, 128, 0, stream>>>(xrt, EH, EH, t, rel, rel, ew, denom,
                                             xr_t, EH, 0, N_EDGE);

  // ---- 7) R->E attention #1 (x_r_h, r2e_*): xh1, xt1 ----
  const int relBlocks = cdiv((long)N_REL * 32, TB);
  float* xh1 = S5;
  float* xt1 = S6;
  k_rowdot2<<<rowBlocks, TB, 0, stream>>>(xe, EH, EH, r2e_ah, r2e_at, sv0, sv1, N_NODES);
  k_rowdot2<<<relBlocks, TB, 0, stream>>>(xr_h, EH, EH, r2e_ar, r2e_ar, sv2, sv3, N_REL);
  soft_edges(sv0, h, sv2, rel, h, N_NODES, N_EDGE);
  k_fill_f32<<<8192, TB, 0, stream>>>(xh1, 0.f, NE300);
  k_scatter_soft<<<N_EDGE, 128, 0, stream>>>(xr_h, EH, EH, rel, h, h, ew, denom,
                                             xh1, EH, 0, N_EDGE);
  soft_edges(sv1, t, sv2, rel, t, N_NODES, N_EDGE);
  k_fill_f32<<<8192, TB, 0, stream>>>(xt1, 0.f, NE300);
  k_scatter_soft<<<N_EDGE, 128, 0, stream>>>(xr_h, EH, EH, rel, t, t, ew, denom,
                                             xt1, EH, 0, N_EDGE);

  // ---- 8) R->E attention #2 (x_r_t, r2et_*): xh2, xt2 ----
  float* xh2 = S1;  // xrh dead
  float* xt2 = S2;  // xrt dead
  k_rowdot2<<<rowBlocks, TB, 0, stream>>>(xe, EH, EH, r2et_ah, r2et_at, sv0, sv1, N_NODES);
  k_rowdot2<<<relBlocks, TB, 0, stream>>>(xr_t, EH, EH, r2et_ar, r2et_ar, sv2, sv3, N_REL);
  soft_edges(sv0, h, sv2, rel, h, N_NODES, N_EDGE);
  k_fill_f32<<<8192, TB, 0, stream>>>(xh2, 0.f, NE300);
  k_scatter_soft<<<N_EDGE, 128, 0, stream>>>(xr_t, EH, EH, rel, h, h, ew, denom,
                                             xh2, EH, 0, N_EDGE);
  soft_edges(sv1, t, sv2, rel, t, N_NODES, N_EDGE);
  k_fill_f32<<<8192, TB, 0, stream>>>(xt2, 0.f, NE300);
  k_scatter_soft<<<N_EDGE, 128, 0, stream>>>(xr_t, EH, EH, rel, t, t, ew, denom,
                                             xt2, EH, 0, N_EDGE);

  // ---- 9) Assemble x_cat into d_out cols [0,900) ----
  k_fill_f32<<<16384, TB, 0, stream>>>(dout, 0.f, (long)N_NODES * LD_OUT);
  k_copy_cols<<<8192, TB, 0, stream>>>(xe, dout, N_NODES, EH, LD_OUT, 0);
  gemm(xh1, hw3_W, S4);
  k_highway<<<8192, TB, 0, stream>>>(S4, hw3_b, xh1, xh2, dout, N_NODES, EH, LD_OUT, EH);
  gemm(xt1, hw4_W, S4);
  k_highway<<<8192, TB, 0, stream>>>(S4, hw4_b, xt1, xt2, dout, N_NODES, EH, LD_OUT, 2 * EH);

  // ---- 10) Final GAT on x_cat -> d_out cols [900,1800), relu ----
  k_rowdot2<<<rowBlocks, TB, 0, stream>>>(dout, LD_OUT, DCAT, gat_ai, gat_aj, sv0, sv1,
                                          N_NODES);
  soft_edges(sv0, dstA, sv1, srcA, dstA, N_NODES, N_EDGEA);
  k_scatter_soft<<<N_EDGEA, 128, 0, stream>>>(dout, LD_OUT, DCAT, srcA, dstA, dstA, ew,
                                              denom, dout, LD_OUT, DCAT, N_EDGEA);
  k_relu_strided<<<16384, TB, 0, stream>>>(dout, N_NODES, DCAT, LD_OUT, DCAT);
}
